// GAT_39814346834506
// MI455X (gfx1250) — compile-verified, matrix operand order checked by reading, not run on hardware
//
#include <hip/hip_runtime.h>

#define LRELU_SLOPE 0.2f

typedef __attribute__((ext_vector_type(16))) __bf16    v16bf;
typedef __attribute__((ext_vector_type(8)))  float     v8f;
typedef __attribute__((ext_vector_type(4)))  unsigned  v4u;

// ---- helpers ---------------------------------------------------------------

// pack two f32 into one dword of 2 bf16 (round-to-nearest-even), lo in [15:0]
__device__ __forceinline__ unsigned pack_bf16x2(float lo, float hi) {
  unsigned a = __float_as_uint(lo);
  unsigned b = __float_as_uint(hi);
  a += 0x7FFFu + ((a >> 16) & 1u);
  b += 0x7FFFu + ((b >> 16) & 1u);
  return (a >> 16) | (b & 0xFFFF0000u);
}

// order-preserving float <-> uint encoding for atomicMax-based segment max
__device__ __forceinline__ unsigned enc_f(float f) {
  unsigned b = __float_as_uint(f);
  return (b & 0x80000000u) ? ~b : (b | 0x80000000u);
}
__device__ __forceinline__ float dec_f(unsigned e) {
  return (e & 0x80000000u) ? __uint_as_float(e & 0x7FFFFFFFu)
                           : __uint_as_float(~e);
}

// ---- streaming f32 -> bf16 conversion (2 elems / thread) -------------------
__global__ void cvt_bf16(const float* __restrict__ in, unsigned* __restrict__ out,
                         int n2)  // n2 = element_count / 2
{
  int i = blockIdx.x * blockDim.x + threadIdx.x;
  if (i < n2) {
    float2 f = ((const float2*)in)[i];
    out[i] = pack_bf16x2(f.x, f.y);
  }
}

// ---- WMMA GEMM: C[n_rows, n_cols] = A[n_rows, K] * W[n_cols, K]^T ----------
// A, W pre-converted to bf16 (row-major along K). One 16x16 C tile per wave32.
// Per K-step: lane's v16bf fragment = two contiguous 16B runs -> 2x b128 loads
// per matrix, then one v_wmma_f32_16x16x32_bf16. No per-step conversion VALU.
// blockDim.x = 32 * (n_cols/16); gridDim.x = ceil(n_rows/16).
__global__ void gemm_bf16_wmma(const unsigned short* __restrict__ A,
                               const unsigned short* __restrict__ W,
                               float* __restrict__ C,
                               int n_rows, int K, int n_cols)
{
  const int lane = threadIdx.x & 31;
  const int wave = threadIdx.x >> 5;
  const int row0 = blockIdx.x << 4;
  const int col0 = wave << 4;
  const int half = lane >> 4;   // K-half select (ISA 7.12.2, 16-bit A 16x32)
  const int rm   = lane & 15;   // M index (A frag) / N index (B frag)

  int arow = row0 + rm;
  if (arow >= n_rows) arow = n_rows - 1;          // clamp; stores are guarded
  const unsigned short* ap = A + (size_t)arow * K + (half << 3);
  const unsigned short* bp = W + (size_t)(col0 + rm) * K + (half << 3);

  v8f acc = {};
#pragma unroll 4
  for (int k0 = 0; k0 < K; k0 += 32) {
    union { v16bf v; v4u q[2]; } fa, fb;
    fa.q[0] = *(const v4u*)(ap + k0);        // K = k0+8h .. +7
    fa.q[1] = *(const v4u*)(ap + k0 + 16);   // K = k0+16+8h .. +7
    fb.q[0] = *(const v4u*)(bp + k0);
    fb.q[1] = *(const v4u*)(bp + k0 + 16);
    acc = __builtin_amdgcn_wmma_f32_16x16x32_bf16(false, fa.v, false, fb.v,
                                                  (short)0, acc, false, false);
  }

  // C layout: lanes 0-15 hold N=rm rows M=r; lanes 16-31 rows M=8+r
  if (row0 + 16 <= n_rows) {                      // full tile: unguarded stores
    float* cp = C + (size_t)(row0 + (half << 3)) * n_cols + (col0 + rm);
#pragma unroll
    for (int r = 0; r < 8; ++r) cp[(size_t)r * n_cols] = acc[r];
  } else {                                        // tail tile (1 of 6250)
#pragma unroll
    for (int r = 0; r < 8; ++r) {
      const int row = row0 + r + (half << 3);
      if (row < n_rows) C[(size_t)row * n_cols + (col0 + rm)] = acc[r];
    }
  }
}

// ---- elementwise / attention kernels ---------------------------------------

__global__ void fill_u32(unsigned* __restrict__ p, unsigned v, int n) {
  int i = blockIdx.x * blockDim.x + threadIdx.x;
  if (i < n) p[i] = v;
}

// el[i] = <ft[i*D..], al[h]>, er likewise; i indexes (node, head) flat
__global__ void attn_scores(const float* __restrict__ ft,
                            const float* __restrict__ al,
                            const float* __restrict__ ar,
                            float* __restrict__ el, float* __restrict__ er,
                            int n, int logH, int D)
{
  int i = blockIdx.x * blockDim.x + threadIdx.x;
  if (i >= (n << logH)) return;
  int h = i & ((1 << logH) - 1);
  const float* f  = ft + (size_t)i * D;
  const float* pl = al + h * D;
  const float* pr = ar + h * D;
  float sl = 0.f, sr = 0.f;
  for (int d = 0; d < D; ++d) {
    float v = f[d];
    sl = fmaf(v, pl[d], sl);
    sr = fmaf(v, pr[d], sr);
  }
  el[i] = sl; er[i] = sr;
}

__global__ void edge_max(const float* __restrict__ el, const float* __restrict__ er,
                         const int* __restrict__ src, const int* __restrict__ dst,
                         unsigned* __restrict__ m, int E, int logH)
{
  int i = blockIdx.x * blockDim.x + threadIdx.x;
  if (i >= (E << logH)) return;
  int e = i >> logH, h = i & ((1 << logH) - 1);
  float v = el[(src[e] << logH) + h] + er[(dst[e] << logH) + h];
  v = (v >= 0.f) ? v : LRELU_SLOPE * v;
  atomicMax(&m[(dst[e] << logH) + h], enc_f(v));
}

__global__ void edge_sum(const float* __restrict__ el, const float* __restrict__ er,
                         const int* __restrict__ src, const int* __restrict__ dst,
                         const unsigned* __restrict__ m, float* __restrict__ s,
                         int E, int logH)
{
  int i = blockIdx.x * blockDim.x + threadIdx.x;
  if (i >= (E << logH)) return;
  int e = i >> logH, h = i & ((1 << logH) - 1);
  int di = (dst[e] << logH) + h;
  float v = el[(src[e] << logH) + h] + er[di];
  v = (v >= 0.f) ? v : LRELU_SLOPE * v;
  atomicAdd(&s[di], __expf(v - dec_f(m[di])));
}

// one thread per (edge, channel); channel c -> head h = c >> logD
__global__ void edge_msg(const float* __restrict__ ft,
                         const float* __restrict__ el, const float* __restrict__ er,
                         const unsigned* __restrict__ m, const float* __restrict__ s,
                         const int* __restrict__ src, const int* __restrict__ dst,
                         float* __restrict__ out,
                         int E, int logC, int logD, int logH)
{
  int t = blockIdx.x * blockDim.x + threadIdx.x;
  int e = t >> logC;
  if (e >= E) return;
  int c = t & ((1 << logC) - 1);
  int h = c >> logD;
  int sN = src[e], dN = dst[e];
  float v = el[(sN << logH) + h] + er[(dN << logH) + h];
  v = (v >= 0.f) ? v : LRELU_SLOPE * v;
  int mi = (dN << logH) + h;
  float a = __expf(v - dec_f(m[mi])) / s[mi];
  atomicAdd(&out[((size_t)dN << logC) + c], ft[((size_t)sN << logC) + c] * a);
}

__global__ void relu_copy(const float* __restrict__ in, float* __restrict__ out, int n) {
  int i = blockIdx.x * blockDim.x + threadIdx.x;
  if (i < n) out[i] = fmaxf(in[i], 0.f);
}

// ---- host-side orchestration -----------------------------------------------

extern "C" void kernel_launch(void* const* d_in, const int* in_sizes, int n_in,
                              void* d_out, int out_size, void* d_ws, size_t ws_size,
                              hipStream_t stream)
{
  const float* feats = (const float*)d_in[0];
  const int*   src   = (const int*)d_in[1];
  const int*   dst   = (const int*)d_in[2];
  const float* W1    = (const float*)d_in[3];
  const float* al1   = (const float*)d_in[4];
  const float* ar1   = (const float*)d_in[5];
  const float* W2    = (const float*)d_in[6];
  const float* al2   = (const float*)d_in[7];
  const float* ar2   = (const float*)d_in[8];

  const int D_IN = 512, H1 = 8, D1 = 16, D2 = 64;
  const int C1 = H1 * D1;  // 128
  const int C2 = D2;       // 64 (H2 = 1)
  const int N = in_sizes[0] / D_IN;
  const int E = in_sizes[1];

  // ---- workspace layout (float units) --------------------------------------
  // persistent:  ft1 | el1 er1 m1 s1 | X
  // X (union):   phase A: feats_bf16 (N*256) + W1_bf16 (32768)
  //              phase B: rst1 (N*128) | ft2 (N*64) | h1_bf16 (N*64) |
  //                       W2_bf16 (4096) | el2 er2 m2 s2 (N*4)
  float*    ft1 = (float*)d_ws;
  float*    el1 = ft1 + (size_t)N * C1;
  float*    er1 = el1 + (size_t)N * H1;
  unsigned* m1  = (unsigned*)(er1 + (size_t)N * H1);
  float*    s1  = (float*)(m1 + (size_t)N * H1);
  float*    X   = s1 + (size_t)N * H1;

  unsigned short* featsb = (unsigned short*)X;                 // N*512 bf16
  unsigned short* W1b    = featsb + (size_t)N * D_IN;          // 128*512 bf16

  float*          rst1 = X;                                    // N*128 f32
  float*          ft2  = rst1 + (size_t)N * C1;                // N*64  f32
  unsigned short* h1b  = (unsigned short*)(ft2 + (size_t)N * C2); // N*128 bf16
  unsigned short* W2b  = h1b + (size_t)N * C1;                 // 64*128 bf16
  float*          el2  = (float*)(W2b + C2 * C1);
  float*          er2  = el2 + N;
  unsigned*       m2   = (unsigned*)(er2 + N);
  float*          s2   = (float*)(m2 + N);

  float* out    = (float*)d_out;
  float* out_h1 = out + (size_t)N * D_IN;
  float* out_h  = out_h1 + (size_t)N * C1;

  // output[0]: straight copy of feats (205 MB, pure HBM stream)
  hipMemcpyAsync(out, feats, (size_t)N * D_IN * sizeof(float),
                 hipMemcpyDeviceToDevice, stream);

  const int mblocks = (N + 15) / 16;

  // ---- layer 1 -------------------------------------------------------------
  { int n2 = N * D_IN / 2;                   // feats -> bf16
    cvt_bf16<<<(n2 + 255) / 256, 256, 0, stream>>>(feats, (unsigned*)featsb, n2); }
  { int n2 = C1 * D_IN / 2;                  // W1 -> bf16
    cvt_bf16<<<(n2 + 255) / 256, 256, 0, stream>>>(W1, (unsigned*)W1b, n2); }
  gemm_bf16_wmma<<<mblocks, 32 * (C1 / 16), 0, stream>>>(featsb, W1b, ft1, N, D_IN, C1);

  { int n = N * (H1 + H1);                   // m1 | s1 contiguous
    fill_u32<<<(n + 255) / 256, 256, 0, stream>>>(m1, 0u, n); }
  attn_scores<<<((N << 3) + 255) / 256, 256, 0, stream>>>(ft1, al1, ar1, el1, er1, N, 3, D1);
  edge_max<<<((E << 3) + 255) / 256, 256, 0, stream>>>(el1, er1, src, dst, m1, E, 3);
  edge_sum<<<((E << 3) + 255) / 256, 256, 0, stream>>>(el1, er1, src, dst, m1, s1, E, 3);

  { int n = N * C1;                          // rst1 zero (aliases dead featsb)
    fill_u32<<<(n + 255) / 256, 256, 0, stream>>>((unsigned*)rst1, 0u, n); }
  edge_msg<<<((E << 7) + 255) / 256, 256, 0, stream>>>(ft1, el1, er1, m1, s1,
                                                       src, dst, rst1, E, 7, 4, 3);
  relu_copy<<<(N * C1 + 255) / 256, 256, 0, stream>>>(rst1, out_h1, N * C1);

  // ---- layer 2 (H2 = 1, mean over heads = identity) ------------------------
  { int n2 = N * C1 / 2;                     // h1 -> bf16
    cvt_bf16<<<(n2 + 255) / 256, 256, 0, stream>>>(out_h1, (unsigned*)h1b, n2); }
  { int n2 = C2 * C1 / 2;                    // W2 -> bf16
    cvt_bf16<<<(n2 + 255) / 256, 256, 0, stream>>>(W2, (unsigned*)W2b, n2); }
  gemm_bf16_wmma<<<mblocks, 32 * (C2 / 16), 0, stream>>>(h1b, W2b, ft2, N, C1, C2);

  { int n = N * 2;                           // m2 | s2 contiguous
    fill_u32<<<(n + 255) / 256, 256, 0, stream>>>(m2, 0u, n); }
  attn_scores<<<(N + 255) / 256, 256, 0, stream>>>(ft2, al2, ar2, el2, er2, N, 0, D2);
  edge_max<<<(E + 255) / 256, 256, 0, stream>>>(el2, er2, src, dst, m2, E, 0);
  edge_sum<<<(E + 255) / 256, 256, 0, stream>>>(el2, er2, src, dst, m2, s2, E, 0);

  { int n = N * C2;                          // out_h zero, then accumulate
    fill_u32<<<(n + 255) / 256, 256, 0, stream>>>((unsigned*)out_h, 0u, n); }
  edge_msg<<<((E << 6) + 255) / 256, 256, 0, stream>>>(ft2, el2, er2, m2, s2,
                                                       src, dst, out_h, E, 6, 6, 0);
}
